// OptimizedMoE_53266184405701
// MI455X (gfx1250) — matrix operand order repaired and checked by reading, not compiled, not run on hardware
//
#include <hip/hip_runtime.h>
#include <hip/hip_bf16.h>

// ---------------------------------------------------------------------------
// MoE top-2 dispatch + grouped GEMM for gfx1250 (wave32, WMMA 16x16x32 bf16)
// Pipeline: zero -> gate/bucket -> per-expert gathered GEMM (bf16 WMMA, fp32 acc)
// ---------------------------------------------------------------------------

typedef __attribute__((ext_vector_type(16))) __bf16       v16bf;
typedef __attribute__((ext_vector_type(2)))  __bf16       v2bf;
typedef __attribute__((ext_vector_type(8)))  float        v8f;
typedef __attribute__((ext_vector_type(4)))  float        v4f;
typedef __attribute__((ext_vector_type(4)))  unsigned int v4u;

#define T_TOK 8192   // B*S tokens
#define DDIM  1024   // model dim
#define NEXP  8      // experts
#define HDIM  1024   // expert out dim
#define BSTRIDE 72   // padded LDS row stride in bf16 elems (rows stay 16B aligned: 144B)

// two fp32 -> packed bf16 dword (v_cvt_pk_bf16_f32 on gfx1250)
__device__ __forceinline__ unsigned pack2(float lo, float hi) {
#if __has_builtin(__builtin_amdgcn_cvt_pk_bf16_f32)
    v2bf p = __builtin_amdgcn_cvt_pk_bf16_f32(lo, hi);
    return __builtin_bit_cast(unsigned, p);
#else
    v2bf p = { (__bf16)lo, (__bf16)hi };
    return __builtin_bit_cast(unsigned, p);
#endif
}

// ---------------------------------------------------------------------------
// Kernel 0: zero output + expert counters (harness poisons d_out/d_ws)
// ---------------------------------------------------------------------------
__global__ void moe_zero_kernel(float* __restrict__ out, int* __restrict__ cnt) {
    size_t i = (size_t)blockIdx.x * blockDim.x + threadIdx.x;
    const size_t n4 = (size_t)T_TOK * HDIM / 4;
    if (i < n4) {
        v4f z = {0.f, 0.f, 0.f, 0.f};
        ((v4f*)out)[i] = z;
    }
    if (i < NEXP) cnt[i] = 0;
}

// ---------------------------------------------------------------------------
// Kernel 1: gating. One wave32 per token: logits = x_t . Wg (E=8),
// top-2, renormalized softmax weight, atomic bucketing into per-expert lists.
// ---------------------------------------------------------------------------
__global__ __launch_bounds__(256)
void moe_gate_kernel(const float* __restrict__ x, const float* __restrict__ Wg,
                     int* __restrict__ cnt, int* __restrict__ tokL,
                     float* __restrict__ gateL) {
    const int wid  = threadIdx.x >> 5;
    const int lane = threadIdx.x & 31;
    const int t    = blockIdx.x * 8 + wid;           // grid = T/8 blocks
    const float* xp = x + (size_t)t * DDIM;

    float a[NEXP];
#pragma unroll
    for (int e = 0; e < NEXP; ++e) a[e] = 0.f;

    for (int d = lane; d < DDIM; d += 32) {
        float xv = xp[d];
        const float* wr = Wg + d * NEXP;             // 8 consecutive floats
#pragma unroll
        for (int e = 0; e < NEXP; ++e) a[e] += xv * wr[e];
    }
#pragma unroll
    for (int off = 16; off > 0; off >>= 1) {
#pragma unroll
        for (int e = 0; e < NEXP; ++e) a[e] += __shfl_xor(a[e], off, 32);
    }

    if (lane == 0) {
        int i0 = 0;
#pragma unroll
        for (int e = 1; e < NEXP; ++e) if (a[e] > a[i0]) i0 = e;
        int i1 = (i0 == 0) ? 1 : 0;
#pragma unroll
        for (int e = 0; e < NEXP; ++e) if (e != i0 && a[e] > a[i1]) i1 = e;
        // renormalized top-2 softmax: denominators cancel
        float g0 = 1.f / (1.f + __expf(a[i1] - a[i0]));
        float g1 = 1.f - g0;
        int p0 = atomicAdd(&cnt[i0], 1);
        tokL[i0 * T_TOK + p0]  = t;
        gateL[i0 * T_TOK + p0] = g0;
        int p1 = atomicAdd(&cnt[i1], 1);
        tokL[i1 * T_TOK + p1]  = t;
        gateL[i1 * T_TOK + p1] = g1;
    }
}

// ---------------------------------------------------------------------------
// Kernel 2: grouped GEMM per expert, WMMA bf16 -> fp32 accumulate.
// grid = (H/128, T/128, E); block = 256 threads = 8 waves.
// Block tile 128(M tokens) x 128(N), K staged 64 at a time; A and B converted
// once per block-stage into LDS (padded stride, b128 traffic).
// Software pipeline: global loads for stage k+1 issued before stage-k WMMAs;
// all fragment ds_loads of a ks-step issued before its 8-WMMA chain.
// Wave tile: 32(M) x 64(N) = 2x4 WMMA 16x16x32 tiles.
// ---------------------------------------------------------------------------
__global__ __launch_bounds__(256)
void moe_gemm_kernel(const float* __restrict__ x,   const float* __restrict__ We,
                     const float* __restrict__ be,  const int* __restrict__ cnt,
                     const int* __restrict__ tokL,  const float* __restrict__ gateL,
                     float* __restrict__ out) {
    const int e  = blockIdx.z;
    const int n0 = blockIdx.x * 128;
    const int m0 = blockIdx.y * 128;
    const int count = cnt[e];
    if (m0 >= count) return;                         // block-uniform early exit

    __shared__ __attribute__((aligned(16))) __bf16 Alds[128 * BSTRIDE]; // 18 KB
    __shared__ __attribute__((aligned(16))) __bf16 Blds[128 * BSTRIDE]; // 18 KB
    __shared__ int   tokS[128];
    __shared__ float gateS[128];

    const int tid = threadIdx.x;
    if (tid < 128) {
        int r = m0 + tid;
        if (r < count) { tokS[tid] = tokL[e * T_TOK + r]; gateS[tid] = gateL[e * T_TOK + r]; }
        else           { tokS[tid] = 0;                   gateS[tid] = 0.f; }
    }
    __syncthreads();

    const int lane = tid & 31;
    const int wid  = tid >> 5;
    const int half = lane >> 4;       // lanes 0-15 vs 16-31
    const int lr   = lane & 15;
    const int wm   = (wid & 3) * 32;  // wave M offset inside block tile
    const int wn   = (wid >> 2) * 64; // wave N offset inside block tile

    v8f acc[2][4];
#pragma unroll
    for (int mi = 0; mi < 2; ++mi)
#pragma unroll
        for (int ni = 0; ni < 4; ++ni) {
            v8f z = {0.f, 0.f, 0.f, 0.f, 0.f, 0.f, 0.f, 0.f};
            acc[mi][ni] = z;
        }

    // --- per-thread staging maps -------------------------------------------
    // A: 2 threads/row, 32 k each (8 x b128 global loads, contiguous)
    const int arow = tid >> 1;
    const int akg  = (tid & 1) * 32;
    const float* aBase = x + (size_t)tokS[arow] * DDIM + akg;
    // B: 2 threads/col, 32 k each (32 coalesced b32 loads, stride H)
    const int bn  = tid & 127;
    const int bkg = (tid >> 7) * 32;
    const float* bBase = We + (size_t)e * DDIM * HDIM + (size_t)bkg * HDIM + (n0 + bn);

    v4f   aPre[8];
    float bPre[32];

    // prefetch stage 0
#pragma unroll
    for (int j = 0; j < 8; ++j) aPre[j] = ((const v4f*)aBase)[j];
#pragma unroll
    for (int kk = 0; kk < 32; ++kk) bPre[kk] = bBase[(size_t)kk * HDIM];

    for (int k0 = 0; k0 < DDIM; k0 += 64) {
        __syncthreads();                 // previous stage's fragment reads done
        // ---- convert (packed) + store staged tiles to LDS as b128 ----------
        {
            v4u* adst = (v4u*)&Alds[arow * BSTRIDE + akg];
#pragma unroll
            for (int j = 0; j < 4; ++j) {
                v4f f0 = aPre[2 * j], f1 = aPre[2 * j + 1];
                v4u w = { pack2(f0.x, f0.y), pack2(f0.z, f0.w),
                          pack2(f1.x, f1.y), pack2(f1.z, f1.w) };
                adst[j] = w;
            }
            v4u* bdst = (v4u*)&Blds[bn * BSTRIDE + bkg];
#pragma unroll
            for (int j = 0; j < 4; ++j) {
                v4u w = { pack2(bPre[8 * j + 0], bPre[8 * j + 1]),
                          pack2(bPre[8 * j + 2], bPre[8 * j + 3]),
                          pack2(bPre[8 * j + 4], bPre[8 * j + 5]),
                          pack2(bPre[8 * j + 6], bPre[8 * j + 7]) };
                bdst[j] = w;
            }
        }
        __syncthreads();

        // ---- issue global prefetch for next stage before the math ----------
        if (k0 + 64 < DDIM) {
            const float* aNext = aBase + (k0 + 64);
#pragma unroll
            for (int j = 0; j < 8; ++j) aPre[j] = ((const v4f*)aNext)[j];
            const float* bNext = bBase + (size_t)(k0 + 64) * HDIM;
#pragma unroll
            for (int kk = 0; kk < 32; ++kk) bPre[kk] = bNext[(size_t)kk * HDIM];
        }

        // ---- WMMA over the 64-wide K stage (two 16x16x32 steps) ------------
#pragma unroll
        for (int ks = 0; ks < 64; ks += 32) {
            // Issue ALL fragment loads first (12 x ds_load_b128 in flight),
            // then run the 8-WMMA chain with staggered dscnt waits.
            union { v4u u[2]; v16bf v; } af[2], bfr[4];
#pragma unroll
            for (int mi = 0; mi < 2; ++mi) {
                const int row = wm + mi * 16 + lr;
                const v4u* ap = (const v4u*)&Alds[row * BSTRIDE + ks + half * 8];
                af[mi].u[0] = ap[0];      // K ks+half*8 .. +7
                af[mi].u[1] = ap[2];      // K ks+16+half*8 .. +7  (+16 elems = +2 v4u)
            }
#pragma unroll
            for (int ni = 0; ni < 4; ++ni) {
                const int nrow = wn + ni * 16 + lr;
                const v4u* bp = (const v4u*)&Blds[nrow * BSTRIDE + ks + half * 8];
                bfr[ni].u[0] = bp[0];
                bfr[ni].u[1] = bp[2];
            }
#pragma unroll
            for (int ni = 0; ni < 4; ++ni)
#pragma unroll
                for (int mi = 0; mi < 2; ++mi)
                    acc[mi][ni] = __builtin_amdgcn_wmma_f32_16x16x32_bf16(
                        false, af[mi].v, false, bfr[ni].v, (short)0, acc[mi][ni],
                        false, false);
        }
    }

    // ---- epilogue: out[t] += g * (acc + bias); two experts can hit the same
    //      token row concurrently -> global f32 atomic adds.
#pragma unroll
    for (int mi = 0; mi < 2; ++mi) {
#pragma unroll
        for (int ni = 0; ni < 4; ++ni) {
            const int col = n0 + wn + ni * 16 + lr;
            const float bias = be[e * HDIM + col];
#pragma unroll
            for (int v = 0; v < 8; ++v) {
                const int rloc = wm + mi * 16 + v + 8 * half;  // C/D layout: M = v + 8*half
                if (m0 + rloc < count) {
                    const int   t = tokS[rloc];
                    const float g = gateS[rloc];
                    atomicAdd(&out[(size_t)t * HDIM + col], g * (acc[mi][ni][v] + bias));
                }
            }
        }
    }
}

// ---------------------------------------------------------------------------
// Host launcher
// ---------------------------------------------------------------------------
extern "C" void kernel_launch(void* const* d_in, const int* in_sizes, int n_in,
                              void* d_out, int out_size, void* d_ws, size_t ws_size,
                              hipStream_t stream) {
    const float* x  = (const float*)d_in[0];   // [B,S,D] = [T,D]
    const float* Wg = (const float*)d_in[1];   // [D,E]
    const float* We = (const float*)d_in[2];   // [E,D,H]
    const float* be = (const float*)d_in[3];   // [E,H]
    float* out = (float*)d_out;                // [T,H]

    // workspace layout: counters | token lists | gate lists  (~512 KB)
    int*   cnt   = (int*)d_ws;
    int*   tokL  = cnt + 16;
    float* gateL = (float*)(tokL + NEXP * T_TOK);

    // 0) zero out + counters
    {
        const size_t n4 = (size_t)T_TOK * HDIM / 4;
        dim3 grid((unsigned)((n4 + 255) / 256));
        moe_zero_kernel<<<grid, 256, 0, stream>>>(out, cnt);
    }
    // 1) gating + bucketing (one wave per token)
    moe_gate_kernel<<<dim3(T_TOK / 8), 256, 0, stream>>>(x, Wg, cnt, tokL, gateL);
    // 2) grouped GEMM; blocks beyond each expert's count exit immediately
    {
        dim3 grid(HDIM / 128, T_TOK / 128, NEXP);   // (8, 64, 8)
        moe_gemm_kernel<<<grid, 256, 0, stream>>>(x, We, be, cnt, tokL, gateL, out);
    }
}
